// MusicAttrCVAE_38139309588582
// MI455X (gfx1250) — matrix operand order repaired and checked by reading, not compiled, not run on hardware
//
#include <hip/hip_runtime.h>
#include <hip/hip_bf16.h>
#include <math.h>

// ---------------------------------------------------------------------------
// Types for CDNA5 WMMA (wave32, v_wmma_f32_16x16x32_bf16)
// ---------------------------------------------------------------------------
typedef __bf16 bf16;
typedef bf16  v16bf __attribute__((ext_vector_type(16)));
typedef bf16  v8bf  __attribute__((ext_vector_type(8)));
typedef float v8f   __attribute__((ext_vector_type(8)));

#define WMMA_BF16(a, b, c) \
  __builtin_amdgcn_wmma_f32_16x16x32_bf16(false, (a), false, (b), (short)0, (c), false, false)

// A-fragment (16x32 bf16, MxK): lane L holds row M = L&15.
// Per ISA 7.12.2: lanes 0-15 hold K {0..7,16..23}, lanes 16-31 hold K {8..15,24..31}.
__device__ __forceinline__ v16bf load_fragA(const bf16* p, int lane) {
  const int off = (lane >> 4) << 3;                 // 0 or 8
  v8bf lo = *(const v8bf*)(p + off);                // K = off .. off+7
  v8bf hi = *(const v8bf*)(p + off + 16);           // K = off+16 .. off+23
  return __builtin_shufflevector(lo, hi, 0,1,2,3,4,5,6,7,8,9,10,11,12,13,14,15);
}
// B-fragment (32x16 bf16, KxN), weights stored [N, K] row-major:
// lane L holds column N = L&15; lanes 0-15: K {0..7,16..23}, lanes 16-31: K {8..15,24..31}.
__device__ __forceinline__ v16bf load_fragB(const bf16* p, int lane) {
  const int off = (lane >> 4) << 3;                 // 0 or 8
  v8bf lo = *(const v8bf*)(p + off);
  v8bf hi = *(const v8bf*)(p + off + 16);
  return __builtin_shufflevector(lo, hi, 0,1,2,3,4,5,6,7,8,9,10,11,12,13,14,15);
}

__device__ __forceinline__ float sigmoidf_(float x) { return 1.0f / (1.0f + expf(-x)); }

// ---------------------------------------------------------------------------
// Generic WMMA GEMM:  C[M,N] = A[M,K](bf16) * W[N,K]^T(bf16) + bias[N]
// Block = 128 threads = 4 waves; each wave computes a 16(M) x 64(N) strip.
// Requires: M % 64 == 0, N % 64 == 0, K % 32 == 0.
// ---------------------------------------------------------------------------
__global__ void wmma_gemm_bf16(const bf16* __restrict__ A, const bf16* __restrict__ Bw,
                               const float* __restrict__ bias, float* __restrict__ C,
                               int M, int N, int K)
{
  const int wave  = threadIdx.x >> 5;
  const int lane  = threadIdx.x & 31;
  const int mBase = blockIdx.y * 64 + wave * 16;
  const int nBase = blockIdx.x * 64;
  if (mBase >= M) return;

  v8f acc0 = {}, acc1 = {}, acc2 = {}, acc3 = {};
  const bf16* aRow  = A  + (size_t)(mBase + (lane & 15)) * K;
  const bf16* bRow  = Bw + (size_t)(nBase + (lane & 15)) * K;
  const size_t bstep = (size_t)16 * K;

  for (int k = 0; k < K; k += 32) {
    v16bf a  = load_fragA(aRow + k, lane);
    v16bf b0 = load_fragB(bRow + k, lane);
    v16bf b1 = load_fragB(bRow + bstep + k, lane);
    v16bf b2 = load_fragB(bRow + 2 * bstep + k, lane);
    v16bf b3 = load_fragB(bRow + 3 * bstep + k, lane);
    acc0 = WMMA_BF16(a, b0, acc0);
    acc1 = WMMA_BF16(a, b1, acc1);
    acc2 = WMMA_BF16(a, b2, acc2);
    acc3 = WMMA_BF16(a, b3, acc3);
  }

  const int mRow = mBase + ((lane >> 4) << 3);
  const int nCol = nBase + (lane & 15);
  v8f accs[4] = {acc0, acc1, acc2, acc3};
#pragma unroll
  for (int j = 0; j < 4; ++j) {
    const int n = nCol + j * 16;
    const float bv = bias ? bias[n] : 0.0f;
#pragma unroll
    for (int v = 0; v < 8; ++v)
      C[(size_t)(mRow + v) * N + n] = accs[j][v] + bv;
  }
}

// ---------------------------------------------------------------------------
// Fused GRU step (wide): gh = hA @ Whh^T (+bhh in epilogue), gi precomputed.
// 512 threads = 16 waves/block; each wave owns ONE 16x16 tile of all 3 gates
// (r@n, z@H+n, n@2H+n) -> 256 waves across the grid, WMMA chain of 3*K/32.
// gridDim.z selects between two independent problems (fwd/bwd encoder merge).
//   r  = sigmoid(gi_r + gh_r);  z = sigmoid(gi_z + gh_z)
//   nn = tanh(gi_n + r*gh_n);   h = (1-z)*nn + z*h_prev
// hA (bf16 A-operand) must differ from hbf (ping-pong: every block reads all
// K columns of hA). hprev/hout fp32 may alias (1:1 element ownership).
// ---------------------------------------------------------------------------
struct GruPtrs {
  const bf16*  hA;     // [Bn, H] bf16 h_prev (A operand)
  const bf16*  Whh;    // [3H, H] bf16
  const float* bhh;    // [3H]
  const float* gi;     // [Bn, 3H] precomputed input gates (incl. bih)
  const float* hprev;  // [Bn, H] fp32 h_prev
  float*       hout;   // [Bn, H] fp32 h_new
  bf16*        hbf;    // [Bn, H] bf16 h_new
};

__global__ void gru_step(GruPtrs p0, GruPtrs p1, int Bn, int H)
{
  const GruPtrs p = (blockIdx.z == 1) ? p1 : p0;
  const int wave  = threadIdx.x >> 5;
  const int lane  = threadIdx.x & 31;
  const int mBase = blockIdx.y * 64 + (wave & 3) * 16;
  const int nBase = blockIdx.x * 64 + (wave >> 2) * 16;
  const int H3 = 3 * H;

  v8f aR = {}, aZ = {}, aN = {};
  const bf16* aRow = p.hA + (size_t)(mBase + (lane & 15)) * H;
  const int nLane = nBase + (lane & 15);
  const bf16* bR = p.Whh + (size_t)nLane * H;
  const bf16* bZ = p.Whh + (size_t)(H + nLane) * H;
  const bf16* bN = p.Whh + (size_t)(2 * H + nLane) * H;

  for (int k = 0; k < H; k += 32) {
    v16bf a  = load_fragA(aRow + k, lane);
    v16bf br = load_fragB(bR + k, lane);
    v16bf bz = load_fragB(bZ + k, lane);
    v16bf bn = load_fragB(bN + k, lane);
    aR = WMMA_BF16(a, br, aR);
    aZ = WMMA_BF16(a, bz, aZ);
    aN = WMMA_BF16(a, bn, aN);
  }

  const int mRow = mBase + ((lane >> 4) << 3);
  const int n = nBase + (lane & 15);
  const float br_ = p.bhh[n], bz_ = p.bhh[H + n], bn_ = p.bhh[2 * H + n];
#pragma unroll
  for (int v = 0; v < 8; ++v) {
    const int m = mRow + v;
    const float* gim = p.gi + (size_t)m * H3;
    float r  = sigmoidf_(gim[n] + aR[v] + br_);
    float z  = sigmoidf_(gim[H + n] + aZ[v] + bz_);
    float nn = tanhf(gim[2 * H + n] + r * (aN[v] + bn_));
    float h  = (1.0f - z) * nn + z * p.hprev[(size_t)m * H + n];
    p.hout[(size_t)m * H + n] = h;
    p.hbf[(size_t)m * H + n]  = (bf16)h;
  }
}

// ---------------------------------------------------------------------------
// Fused dual-GEMM GRU step (decoder layer 2), wide 16-wave version:
//   gi = xA @ Wih^T + bih ; gh = hA @ Whh^T + bhh, all in-register.
// r/z gates: gi+gh share one accumulator; n gate needs split accumulators.
// ---------------------------------------------------------------------------
__global__ void gru_step_dual(const bf16* __restrict__ xA, const bf16* __restrict__ hA,
                              const bf16* __restrict__ Wih, const bf16* __restrict__ Whh,
                              const float* __restrict__ bih, const float* __restrict__ bhh,
                              const float* __restrict__ hprev, float* __restrict__ hout,
                              bf16* __restrict__ hbf, int Bn, int H)
{
  const int wave  = threadIdx.x >> 5;
  const int lane  = threadIdx.x & 31;
  const int mBase = blockIdx.y * 64 + (wave & 3) * 16;
  const int nBase = blockIdx.x * 64 + (wave >> 2) * 16;

  v8f aR = {}, aZ = {}, aGiN = {}, aGhN = {};
  const bf16* xRow = xA + (size_t)(mBase + (lane & 15)) * H;
  const bf16* hRow = hA + (size_t)(mBase + (lane & 15)) * H;
  const int nLane = nBase + (lane & 15);
  const bf16* iR = Wih + (size_t)nLane * H;
  const bf16* iZ = Wih + (size_t)(H + nLane) * H;
  const bf16* iN = Wih + (size_t)(2 * H + nLane) * H;
  const bf16* hR = Whh + (size_t)nLane * H;
  const bf16* hZ = Whh + (size_t)(H + nLane) * H;
  const bf16* hN = Whh + (size_t)(2 * H + nLane) * H;

  for (int k = 0; k < H; k += 32) {
    v16bf ax = load_fragA(xRow + k, lane);
    v16bf ah = load_fragA(hRow + k, lane);
    aR   = WMMA_BF16(ax, load_fragB(iR + k, lane), aR);
    aR   = WMMA_BF16(ah, load_fragB(hR + k, lane), aR);
    aZ   = WMMA_BF16(ax, load_fragB(iZ + k, lane), aZ);
    aZ   = WMMA_BF16(ah, load_fragB(hZ + k, lane), aZ);
    aGiN = WMMA_BF16(ax, load_fragB(iN + k, lane), aGiN);
    aGhN = WMMA_BF16(ah, load_fragB(hN + k, lane), aGhN);
  }

  const int mRow = mBase + ((lane >> 4) << 3);
  const int n = nBase + (lane & 15);
  const float brz = bih[n] + bhh[n];
  const float bzz = bih[H + n] + bhh[H + n];
  const float bin_ = bih[2 * H + n], bhn_ = bhh[2 * H + n];
#pragma unroll
  for (int v = 0; v < 8; ++v) {
    const int m = mRow + v;
    float r  = sigmoidf_(aR[v] + brz);
    float z  = sigmoidf_(aZ[v] + bzz);
    float nn = tanhf(aGiN[v] + bin_ + r * (aGhN[v] + bhn_));
    float h  = (1.0f - z) * nn + z * hprev[(size_t)m * H + n];
    hout[(size_t)m * H + n] = h;
    hbf[(size_t)m * H + n]  = (bf16)h;
  }
}

// ---------------------------------------------------------------------------
// Elementwise helpers
// ---------------------------------------------------------------------------
__global__ void convert_weight(const float* __restrict__ src, bf16* __restrict__ dst,
                               int Nsrc, int Ksrc, int Ndst, int Kdst)
{
  int idx = blockIdx.x * blockDim.x + threadIdx.x;
  if (idx >= Ndst * Kdst) return;
  int n = idx / Kdst, k = idx - n * Kdst;
  float v = (n < Nsrc && k < Ksrc) ? src[(size_t)n * Ksrc + k] : 0.0f;
  dst[idx] = (bf16)v;
}

__global__ void pad_bias_f32(const float* __restrict__ src, float* __restrict__ dst,
                             int Nsrc, int Ndst)
{
  int i = blockIdx.x * blockDim.x + threadIdx.x;
  if (i < Ndst) dst[i] = (i < Nsrc) ? src[i] : 0.0f;
}

__global__ void zero_f32(float* p, int n) {
  int i = blockIdx.x * blockDim.x + threadIdx.x;
  if (i < n) p[i] = 0.0f;
}
__global__ void zero_bf16(bf16* p, int n) {
  int i = blockIdx.x * blockDim.x + threadIdx.x;
  if (i < n) p[i] = (bf16)0.0f;
}
__global__ void cvt_f2b(const float* __restrict__ s, bf16* __restrict__ d, int n) {
  int i = blockIdx.x * blockDim.x + threadIdx.x;
  if (i < n) d[i] = (bf16)s[i];
}

// xs[t][b][k] (bf16, Kp=352): [x(342) | r_density | n_density | 0-pad]
__global__ void build_xs(const float* __restrict__ x, const float* __restrict__ rd,
                         const float* __restrict__ nd, bf16* __restrict__ xs,
                         int T, int Bn, int R, int Kp)
{
  int idx = blockIdx.x * blockDim.x + threadIdx.x;
  if (idx >= T * Bn * Kp) return;
  int k = idx % Kp; int tb = idx / Kp; int b = tb % Bn; int t = tb / Bn;
  float v = 0.0f;
  if (k < R)           v = x[((size_t)b * T + t) * R + k];
  else if (k == R)     v = rd[b];
  else if (k == R + 1) v = nd[b];
  xs[idx] = (bf16)v;
}

// cin[t][b][k] (bf16, Kp=608): [teacher(342) | zc(258) | 0-pad]
__global__ void build_cin(const float* __restrict__ x, const float* __restrict__ zc,
                          bf16* __restrict__ cin, int T, int Bn, int R, int Zc, int Kp)
{
  int idx = blockIdx.x * blockDim.x + threadIdx.x;
  if (idx >= T * Bn * Kp) return;
  int k = idx % Kp; int tb = idx / Kp; int b = tb % Bn; int t = tb / Bn;
  float v = 0.0f;
  if (k < R) {
    v = (t == 0) ? ((k == R - 1) ? 1.0f : 0.0f)
                 : x[((size_t)b * T + (t - 1)) * R + k];
  } else if (k < R + Zc) {
    v = zc[(size_t)b * Zc + (k - R)];
  }
  cin[idx] = (bf16)v;
}

__global__ void concat_h(const float* __restrict__ hf, const float* __restrict__ hb,
                         bf16* __restrict__ hcat, int Bn, int H)
{
  int idx = blockIdx.x * blockDim.x + threadIdx.x;
  if (idx >= Bn * 2 * H) return;
  int b = idx / (2 * H), j = idx - b * (2 * H);
  float v = (j < H) ? hf[(size_t)b * H + j] : hb[(size_t)b * H + (j - H)];
  hcat[idx] = (bf16)v;
}

// stddev = exp(lvar); z = mu + stddev*eps; zc = [z | rd | nd]
__global__ void reparam(const float* __restrict__ mu, const float* __restrict__ lvar,
                        const float* __restrict__ eps, const float* __restrict__ rd,
                        const float* __restrict__ nd,
                        float* __restrict__ out_mu, float* __restrict__ out_sd,
                        float* __restrict__ out_zc, float* __restrict__ zcf,
                        bf16* __restrict__ zcb, int Bn, int Z, int Zp)
{
  int idx = blockIdx.x * blockDim.x + threadIdx.x;
  if (idx >= Bn * Zp) return;
  int b = idx / Zp, k = idx - b * Zp;
  const int Zc = Z + 2;
  if (k < Z) {
    float m  = mu[(size_t)b * Z + k];
    float sd = expf(lvar[(size_t)b * Z + k]);
    float zz = m + sd * eps[(size_t)b * Z + k];
    out_mu[(size_t)b * Z + k]  = m;
    out_sd[(size_t)b * Z + k]  = sd;
    out_zc[(size_t)b * Zc + k] = zz;
    zcf[(size_t)b * Zc + k]    = zz;
    zcb[idx] = (bf16)zz;
  } else if (k == Z) {
    float v = rd[b];
    out_zc[(size_t)b * Zc + Z] = v; zcf[(size_t)b * Zc + Z] = v; zcb[idx] = (bf16)v;
  } else if (k == Z + 1) {
    float v = nd[b];
    out_zc[(size_t)b * Zc + Z + 1] = v; zcf[(size_t)b * Zc + Z + 1] = v; zcb[idx] = (bf16)v;
  } else {
    zcb[idx] = (bf16)0.0f;
  }
}

// Row-wise log_softmax; logits row = t*Bn + b; output transposed to out[b][t][r].
__global__ void logsoftmax_out(const float* __restrict__ logits, float* __restrict__ out,
                               int T, int Bn, int R, int ldL)
{
  const int row = blockIdx.x;
  const int t = row / Bn, b = row - t * Bn;
  const float* lp = logits + (size_t)row * ldL;
  __shared__ float red[128];
  float mx = -INFINITY;
  for (int i = threadIdx.x; i < R; i += blockDim.x) mx = fmaxf(mx, lp[i]);
  red[threadIdx.x] = mx; __syncthreads();
  for (int s = blockDim.x / 2; s > 0; s >>= 1) {
    if (threadIdx.x < s) red[threadIdx.x] = fmaxf(red[threadIdx.x], red[threadIdx.x + s]);
    __syncthreads();
  }
  mx = red[0]; __syncthreads();
  float sum = 0.0f;
  for (int i = threadIdx.x; i < R; i += blockDim.x) sum += expf(lp[i] - mx);
  red[threadIdx.x] = sum; __syncthreads();
  for (int s = blockDim.x / 2; s > 0; s >>= 1) {
    if (threadIdx.x < s) red[threadIdx.x] += red[threadIdx.x + s];
    __syncthreads();
  }
  const float lse = mx + logf(red[0]);
  float* op = out + ((size_t)b * T + t) * R;
  for (int i = threadIdx.x; i < R; i += blockDim.x) op[i] = lp[i] - lse;
}

// ---------------------------------------------------------------------------
// Host orchestration
// ---------------------------------------------------------------------------
static inline size_t alignup(size_t x) { return (x + 255) & ~(size_t)255; }

extern "C" void kernel_launch(void* const* d_in, const int* in_sizes, int n_in,
                              void* d_out, int out_size, void* d_ws, size_t ws_size,
                              hipStream_t stream)
{
  (void)in_sizes; (void)n_in; (void)out_size; (void)ws_size;
  constexpr int Bn = 64, T = 256, R = 342, H = 1024, Z = 256;
  constexpr int N3H = 3 * H;           // 3072
  constexpr int KX  = 352;             // R+2=344 padded to 32
  constexpr int KC  = 608;             // R+Z+2=600 padded
  constexpr int KZ  = 288;             // Z+2=258 padded
  constexpr int NO  = 384;             // R=342 padded to 64 for output GEMM
  constexpr int H2  = 2 * H;
  const int TB = T * Bn;               // 16384

  const float* x     = (const float*)d_in[0];
  const float* rd    = (const float*)d_in[1];
  const float* nd    = (const float*)d_in[2];
  /* chroma d_in[3] unused by reference */
  const float* eps   = (const float*)d_in[4];
  const float* Wih_f = (const float*)d_in[5];
  const float* Whh_f = (const float*)d_in[6];
  const float* bih_f = (const float*)d_in[7];
  const float* bhh_f = (const float*)d_in[8];
  const float* Wih_b = (const float*)d_in[9];
  const float* Whh_b = (const float*)d_in[10];
  const float* bih_b = (const float*)d_in[11];
  const float* bhh_b = (const float*)d_in[12];
  const float* W_mu  = (const float*)d_in[13];
  const float* b_mu  = (const float*)d_in[14];
  const float* W_var = (const float*)d_in[15];
  const float* b_var = (const float*)d_in[16];
  const float* W_ini = (const float*)d_in[17];
  const float* b_ini = (const float*)d_in[18];
  const float* Wih_g = (const float*)d_in[19];
  const float* Whh_g = (const float*)d_in[20];
  const float* bih_g = (const float*)d_in[21];
  const float* bhh_g = (const float*)d_in[22];
  const float* Wih_g2= (const float*)d_in[23];
  const float* Whh_g2= (const float*)d_in[24];
  const float* bih_g2= (const float*)d_in[25];
  const float* bhh_g2= (const float*)d_in[26];
  const float* W_out = (const float*)d_in[27];
  const float* b_out = (const float*)d_in[28];

  float* out_logp = (float*)d_out;                                  // [B,T,R]
  float* out_mu   = out_logp + (size_t)Bn * T * R;                  // [B,Z]
  float* out_sd   = out_mu + (size_t)Bn * Z;                        // [B,Z]
  float* out_zc   = out_sd + (size_t)Bn * Z;                        // [B,Z+2]

  // ---- workspace carve ----
  char* wp = (char*)d_ws;
  auto take = [&](size_t bytes) -> char* { char* r = wp; wp += alignup(bytes); return r; };
  bf16* wWihF = (bf16*)take((size_t)N3H * KX * 2);
  bf16* wWihB = (bf16*)take((size_t)N3H * KX * 2);
  bf16* wWhhF = (bf16*)take((size_t)N3H * H * 2);
  bf16* wWhhB = (bf16*)take((size_t)N3H * H * 2);
  bf16* wWhhG = (bf16*)take((size_t)N3H * H * 2);
  bf16* wWihG2= (bf16*)take((size_t)N3H * H * 2);
  bf16* wWhhG2= (bf16*)take((size_t)N3H * H * 2);
  bf16* wWihG = (bf16*)take((size_t)N3H * KC * 2);
  bf16* wWmu  = (bf16*)take((size_t)Z * H2 * 2);
  bf16* wWvar = (bf16*)take((size_t)Z * H2 * 2);
  bf16* wWini = (bf16*)take((size_t)H * KZ * 2);
  bf16* wWout = (bf16*)take((size_t)NO * H * 2);
  float* bOutP= (float*)take((size_t)NO * 4);
  bf16* xsB   = (bf16*)take((size_t)TB * KX * 2);
  bf16* cinB  = (bf16*)take((size_t)TB * KC * 2);
  float* giF  = (float*)take((size_t)TB * N3H * 4);   // reused as giG in phase 4+
  float* giB  = (float*)take((size_t)TB * N3H * 4);   // reused as logits in phase 6
  float* hF   = (float*)take((size_t)Bn * H * 4);
  float* hB   = (float*)take((size_t)Bn * H * 4);
  float* h0   = (float*)take((size_t)Bn * H * 4);
  float* h1   = (float*)take((size_t)Bn * H * 4);
  bf16* hFb0  = (bf16*)take((size_t)Bn * H * 2);      // ping-pong bf16 h buffers
  bf16* hFb1  = (bf16*)take((size_t)Bn * H * 2);
  bf16* hBb0  = (bf16*)take((size_t)Bn * H * 2);
  bf16* hBb1  = (bf16*)take((size_t)Bn * H * 2);
  bf16* h0b0  = (bf16*)take((size_t)Bn * H * 2);
  bf16* h0b1  = (bf16*)take((size_t)Bn * H * 2);
  bf16* hcat  = (bf16*)take((size_t)Bn * H2 * 2);
  float* muB  = (float*)take((size_t)Bn * Z * 4);
  float* lvB  = (float*)take((size_t)Bn * Z * 4);
  float* zcF  = (float*)take((size_t)Bn * (Z + 2) * 4);
  bf16* zcB   = (bf16*)take((size_t)Bn * KZ * 2);
  bf16* h1all = (bf16*)take((size_t)TB * H * 2);
  float* giG  = giF;                                   // alias: giF dead after encoder
  float* logit= giB;                                   // alias: giB dead after encoder

  auto ew = [&](int n) { return dim3((n + 255) / 256); };
  auto gemm = [&](const bf16* A, const bf16* Bw, const float* bias, float* C,
                  int M, int N, int K) {
    wmma_gemm_bf16<<<dim3(N / 64, M / 64), 128, 0, stream>>>(A, Bw, bias, C, M, N, K);
  };

  // ---- Phase 0: weight conversion (fp32 -> bf16, K/N padded) ----
  auto cw = [&](const float* s, bf16* d, int Ns, int Ks, int Nd, int Kd) {
    convert_weight<<<ew(Nd * Kd), 256, 0, stream>>>(s, d, Ns, Ks, Nd, Kd);
  };
  cw(Wih_f, wWihF, N3H, R + 2, N3H, KX);
  cw(Wih_b, wWihB, N3H, R + 2, N3H, KX);
  cw(Whh_f, wWhhF, N3H, H, N3H, H);
  cw(Whh_b, wWhhB, N3H, H, N3H, H);
  cw(Whh_g, wWhhG, N3H, H, N3H, H);
  cw(Wih_g2, wWihG2, N3H, H, N3H, H);
  cw(Whh_g2, wWhhG2, N3H, H, N3H, H);
  cw(Wih_g, wWihG, N3H, R + Z + 2, N3H, KC);
  cw(W_mu, wWmu, Z, H2, Z, H2);
  cw(W_var, wWvar, Z, H2, Z, H2);
  cw(W_ini, wWini, H, Z + 2, H, KZ);
  cw(W_out, wWout, R, H, NO, H);
  pad_bias_f32<<<ew(NO), 256, 0, stream>>>(b_out, bOutP, R, NO);

  // ---- Phase 1: encoder input GEMMs (precompute gi for all T, incl. bih) ----
  build_xs<<<ew(TB * KX), 256, 0, stream>>>(x, rd, nd, xsB, T, Bn, R, KX);
  gemm(xsB, wWihF, bih_f, giF, TB, N3H, KX);
  gemm(xsB, wWihB, bih_b, giB, TB, N3H, KX);

  // ---- Phase 2: bidirectional encoder recurrence ----
  // One launch per step covers BOTH directions via gridDim.z = 2.
  zero_f32<<<ew(Bn * H), 256, 0, stream>>>(hF, Bn * H);
  zero_f32<<<ew(Bn * H), 256, 0, stream>>>(hB, Bn * H);
  zero_bf16<<<ew(Bn * H), 256, 0, stream>>>(hFb0, Bn * H);
  zero_bf16<<<ew(Bn * H), 256, 0, stream>>>(hBb0, Bn * H);
  for (int t = 0; t < T; ++t) {
    bf16* fIn  = (t & 1) ? hFb1 : hFb0;
    bf16* fOut = (t & 1) ? hFb0 : hFb1;
    bf16* bIn  = (t & 1) ? hBb1 : hBb0;
    bf16* bOut = (t & 1) ? hBb0 : hBb1;
    GruPtrs pF = { fIn, wWhhF, bhh_f, giF + (size_t)t * Bn * N3H, hF, hF, fOut };
    GruPtrs pB = { bIn, wWhhB, bhh_b, giB + (size_t)(T - 1 - t) * Bn * N3H, hB, hB, bOut };
    gru_step<<<dim3(H / 64, Bn / 64, 2), 512, 0, stream>>>(pF, pB, Bn, H);
  }

  // ---- Phase 3: latent head ----
  concat_h<<<ew(Bn * H2), 256, 0, stream>>>(hF, hB, hcat, Bn, H);
  gemm(hcat, wWmu, b_mu, muB, Bn, Z, H2);
  gemm(hcat, wWvar, b_var, lvB, Bn, Z, H2);
  reparam<<<ew(Bn * KZ), 256, 0, stream>>>(muB, lvB, eps, rd, nd,
                                           out_mu, out_sd, out_zc, zcF, zcB, Bn, Z, KZ);

  // ---- Phase 4: decoder input GEMM (teacher forcing => precomputable) ----
  build_cin<<<ew(TB * KC), 256, 0, stream>>>(x, zcF, cinB, T, Bn, R, Z + 2, KC);
  gemm(cinB, wWihG, bih_g, giG, TB, N3H, KC);
  gemm(zcB, wWini, b_ini, h0, Bn, H, KZ);
  cvt_f2b<<<ew(Bn * H), 256, 0, stream>>>(h0, h0b0, Bn * H);

  // ---- Phase 5: 2-layer decoder recurrence (fused GRU steps) ----
  for (int t = 0; t < T; ++t) {
    bf16* l1In  = (t & 1) ? h0b1 : h0b0;
    bf16* l1Out = (t & 1) ? h0b0 : h0b1;
    // layer 1: gi precomputed, gh fused; h0 <- h0n
    GruPtrs pG = { l1In, wWhhG, bhh_g, giG + (size_t)t * Bn * N3H, h0, h0, l1Out };
    gru_step<<<dim3(H / 64, Bn / 64, 1), 512, 0, stream>>>(pG, pG, Bn, H);
    // layer 2: input + recurrent GEMMs fused (h1_in = h0n at t==0)
    const bf16* h1inB = (t == 0) ? l1Out : (h1all + (size_t)(t - 1) * Bn * H);
    const float* h1inF = (t == 0) ? h0 : h1;
    gru_step_dual<<<dim3(H / 64, Bn / 64, 1), 512, 0, stream>>>(
        l1Out, h1inB, wWihG2, wWhhG2, bih_g2, bhh_g2,
        h1inF, h1, h1all + (size_t)t * Bn * H, Bn, H);
  }

  // ---- Phase 6: output projection + log_softmax (all steps at once) ----
  gemm(h1all, wWout, bOutP, logit, TB, NO, H);
  logsoftmax_out<<<dim3(TB), 128, 0, stream>>>(logit, out_logp, T, Bn, R, NO);
}
